// AMKPDModel_19902878450348
// MI455X (gfx1250) — matrix operand order repaired and checked by reading, not compiled, not run on hardware
//
#include <hip/hip_runtime.h>
#include <hip/hip_bf16.h>
#include <math.h>

#define B_     2
#define N_     1024
#define D_     512
#define H_     8
#define DH_    64
#define L_     4
#define INNER_ 1536
#define V_     32000
#define KS_    3
#define T_     (B_ * N_)
#define CYC_   2   // HCYC*LCYC

typedef __attribute__((ext_vector_type(16))) __bf16 v16bf;
typedef __attribute__((ext_vector_type(8)))  float  v8f;

// ---------------------------------------------------------------- helpers

__device__ __forceinline__ float wave_sum(float x) {
#pragma unroll
    for (int off = 16; off > 0; off >>= 1) x += __shfl_xor(x, off, 32);
    return x;
}

__device__ __forceinline__ float silu(float x) {
    return x / (1.0f + __expf(-x));
}

// A fragment: 16x32 (MxK) bf16, contiguous-K order (same as B frag).
// lane l: M = l&15 ; element e: K = e + 16*(l>=16)
__device__ __forceinline__ v16bf load_a_frag_bf(const __bf16* __restrict__ A, int lda,
                                                int lane, int k0) {
    const int m     = lane & 15;
    const int kbase = k0 + ((lane >> 4) << 4);
    const __bf16* p = A + (size_t)m * lda + kbase;
    v16bf f;
#pragma unroll
    for (int e = 0; e < 16; ++e) f[e] = p[e];
    return f;
}

// B fragment: 32x16 (KxN) bf16, memory N-major with contiguous K (row-major W[n][k]).
// lane l: N = l&15 ; element e: K = e + 16*(l>=16)
__device__ __forceinline__ v16bf load_b_frag_bf(const __bf16* __restrict__ Bm, int ldb,
                                                int lane, int k0) {
    const int n     = lane & 15;
    const int kbase = k0 + ((lane >> 4) << 4);
    const __bf16* p = Bm + (size_t)n * ldb + kbase;
    v16bf f;
#pragma unroll
    for (int e = 0; e < 16; ++e) f[e] = p[e];
    return f;
}

// ---------------------------------------------------------------- kernels

__global__ void k_f2bf(const float* __restrict__ in, __bf16* __restrict__ out, int n) {
    const int i = blockIdx.x * 256 + threadIdx.x;
    if (i < n) out[i] = (__bf16)in[i];
}

// X = embedding[ids] * sqrt(D) ; hidden = broadcast(init_hidden)
__global__ void k_embed(const int* __restrict__ ids, const float* __restrict__ emb,
                        const float* __restrict__ ih, float* __restrict__ X,
                        float* __restrict__ hid) {
    const size_t i = (size_t)blockIdx.x * blockDim.x + threadIdx.x;
    if (i >= (size_t)T_ * D_) return;
    const int t = (int)(i / D_);
    const int d = (int)(i % D_);
    X[i]   = emb[(size_t)ids[t] * D_ + d] * 22.627416997969522f; // sqrt(512)
    hid[i] = ih[d];
}

__global__ void k_ropecache(float* __restrict__ cosb, float* __restrict__ sinb) {
    const int i = blockIdx.x * blockDim.x + threadIdx.x;
    if (i >= N_ * DH_) return;
    const int n = i / DH_;
    const int d = i % DH_;
    const int j = d & (DH_ / 2 - 1);
    const float freq = __expf(-logf(10000.0f) * (float)(2 * j) / (float)DH_);
    const float a = (float)n * freq;
    cosb[i] = cosf(a);
    sinb[i] = sinf(a);
}

__global__ void k_add(float* __restrict__ dst, const float* __restrict__ src, int n) {
    const int i = blockIdx.x * blockDim.x + threadIdx.x;
    if (i < n) dst[i] += src[i];
}

// C[M,Ncols] = A[M,K] @ W[Ncols,K]^T, bf16 operands, f32 accumulate.
// Block = 128 threads (4 waves); block tile 16 x 256; wave w owns cols [w*64, w*64+64).
// The 16x32 A tile per K-step is staged into LDS with async global->LDS copies.
__global__ void k_gemm_bf(const __bf16* __restrict__ A, const __bf16* __restrict__ W,
                          float* __restrict__ C, int Ncols, int K) {
    const int tid  = threadIdx.x;
    const int lane = tid & 31;
    const int wv   = tid >> 5;
    const int row0 = blockIdx.y * 16;
    const int colW = blockIdx.x * 256 + wv * 64;

    __shared__ __attribute__((aligned(16))) __bf16 As[16 * 32];

    v8f acc[4] = {};
    const int n    = lane & 15;
    const int m    = lane & 15;
    const int kb8  = (lane >> 4) << 3;  // unused; A-frag also uses kb16
    const int kb16 = (lane >> 4) << 4;  // B-frag K offset per half-wave

    // async staging assignment: thread copies 8B (4 bf16): row sr, chunk sq
    const int sr = tid >> 3;            // 0..15
    const int sq = tid & 7;             // 0..7
    const uint32_t lds_dst = (uint32_t)(uintptr_t)(&As[0]) + (uint32_t)(sr * 64 + sq * 8);
    const __bf16* gA = A + (size_t)(row0 + sr) * K + sq * 4;

    for (int k0 = 0; k0 < K; k0 += 32) {
        // stage A tile (16x32 bf16 = 1KB) via CDNA5 async load to LDS
        asm volatile("global_load_async_to_lds_b64 %0, %1, off"
                     :: "v"(lds_dst), "v"(gA + k0) : "memory");
        asm volatile("s_wait_asynccnt 0x0" ::: "memory");
        __syncthreads();

        v16bf a;
        {
            const __bf16* p = &As[m * 32 + kb16];
#pragma unroll
            for (int e = 0; e < 16; ++e) a[e] = p[e];
        }
        if (k0 + 32 < K)
            __builtin_prefetch(W + (size_t)(colW + n) * K + k0 + 32 + kb16, 0, 1);
#pragma unroll
        for (int cb = 0; cb < 4; ++cb) {
            const __bf16* p = W + (size_t)(colW + cb * 16 + n) * K + k0 + kb16;
            v16bf b;
#pragma unroll
            for (int e = 0; e < 16; ++e) b[e] = p[e];
            acc[cb] = __builtin_amdgcn_wmma_f32_16x16x32_bf16(false, a, false, b,
                                                              (short)0, acc[cb],
                                                              false, false);
        }
        __syncthreads();
    }
    const int mb = (lane >> 4) << 3;
#pragma unroll
    for (int cb = 0; cb < 4; ++cb)
#pragma unroll
        for (int r = 0; r < 8; ++r)
            C[(size_t)(row0 + mb + r) * Ncols + colW + cb * 16 + n] = acc[cb][r];
}

// qkv[T][3][H][DH] -> bf16 q,k (rope+rms), bf16 v and f32 v, all [B][H][N][DH]
__global__ void k_ropeqkv(const float* __restrict__ qkv, const float* __restrict__ cosb,
                          const float* __restrict__ sinb, __bf16* __restrict__ q,
                          __bf16* __restrict__ k, __bf16* __restrict__ v,
                          float* __restrict__ vf) {
    const int lane = threadIdx.x;            // handles d and d+32
    const int nn = blockIdx.x;
    const int h  = blockIdx.y;
    const int b  = blockIdx.z;
    const size_t t     = (size_t)b * N_ + nn;
    const size_t base  = t * (3 * D_) + (size_t)h * DH_;
    const size_t obase = ((size_t)(b * H_ + h) * N_ + nn) * DH_;
    const float c0 = cosb[nn * DH_ + lane];
    const float c1 = cosb[nn * DH_ + lane + 32];
    const float s0 = sinb[nn * DH_ + lane];
    const float s1 = sinb[nn * DH_ + lane + 32];
    {   // q
        const float x0 = qkv[base + lane];
        const float x1 = qkv[base + lane + 32];
        const float y0 = x0 * c0 - x1 * s0;
        const float y1 = x1 * c1 + x0 * s1;
        const float ss = wave_sum(y0 * y0 + y1 * y1);
        const float inv = rsqrtf(ss * (1.0f / DH_) + 1e-5f);
        q[obase + lane]      = (__bf16)(y0 * inv);
        q[obase + lane + 32] = (__bf16)(y1 * inv);
    }
    {   // k
        const float x0 = qkv[base + D_ + lane];
        const float x1 = qkv[base + D_ + lane + 32];
        const float y0 = x0 * c0 - x1 * s0;
        const float y1 = x1 * c1 + x0 * s1;
        const float ss = wave_sum(y0 * y0 + y1 * y1);
        const float inv = rsqrtf(ss * (1.0f / DH_) + 1e-5f);
        k[obase + lane]      = (__bf16)(y0 * inv);
        k[obase + lane + 32] = (__bf16)(y1 * inv);
    }
    {   // v
        const float v0 = qkv[base + 2 * D_ + lane];
        const float v1 = qkv[base + 2 * D_ + lane + 32];
        v[obase + lane]       = (__bf16)v0;
        v[obase + lane + 32]  = (__bf16)v1;
        vf[obase + lane]      = v0;
        vf[obase + lane + 32] = v1;
    }
}

// Flash-style (elu+1)^2 attention: one wave per (b, h, 16-row q tile).
__global__ void k_attn(const __bf16* __restrict__ q, const __bf16* __restrict__ k,
                       const __bf16* __restrict__ v, const float* __restrict__ vf,
                       float* __restrict__ mout) {
    const int lane = threadIdx.x;
    const int qt = blockIdx.x;
    const int h  = blockIdx.y;
    const int b  = blockIdx.z;
    const size_t hb = (size_t)((b * H_ + h) * N_) * DH_;
    const __bf16* qh = q + hb;
    const __bf16* kh = k + hb;
    const __bf16* vh = v + hb;
    const float*  vfh = vf + hb;
    const float scale = 0.125f; // 1/sqrt(DH)

    const v16bf qa0 = load_a_frag_bf(qh + (size_t)qt * 16 * DH_, DH_, lane, 0);
    const v16bf qa1 = load_a_frag_bf(qh + (size_t)qt * 16 * DH_, DH_, lane, 32);

    v8f acc[4] = {};
    float rowsum = 0.0f;       // valid in lanes 0..15 (row = lane)
    __shared__ float Wt[16 * 32];

    const int n  = lane & 15;
    const int mb = (lane >> 4) << 3;

    for (int j = 0; j < N_; j += 32) {
#pragma unroll
        for (int half = 0; half < 2; ++half) {
            const v16bf b0 = load_b_frag_bf(kh + (size_t)(j + half * 16) * DH_, DH_, lane, 0);
            const v16bf b1 = load_b_frag_bf(kh + (size_t)(j + half * 16) * DH_, DH_, lane, 32);
            v8f s = {};
            s = __builtin_amdgcn_wmma_f32_16x16x32_bf16(false, qa0, false, b0,
                                                        (short)0, s, false, false);
            s = __builtin_amdgcn_wmma_f32_16x16x32_bf16(false, qa1, false, b1,
                                                        (short)0, s, false, false);
#pragma unroll
            for (int r = 0; r < 8; ++r) {
                const float x = s[r] * scale;
                const float w = (x > 0.0f) ? (x + 1.0f) : __expf(x); // elu(x)+1
                Wt[(mb + r) * 32 + half * 16 + n] = w * w;
            }
        }
        __syncthreads();
        if (lane < 16) {
            float rs = 0.0f;
#pragma unroll
            for (int c = 0; c < 32; ++c) rs += Wt[lane * 32 + c];
            rowsum += rs;
        }
        // Re-layout W tile: C-layout -> A-fragment via LDS
        v16bf wa;
        {
            const int kb2 = (lane >> 4) << 4;
            const float* p = &Wt[(lane & 15) * 32 + kb2];
#pragma unroll
            for (int e = 0; e < 16; ++e) wa[e] = (__bf16)p[e];
        }
        // v B-fragments: memory is key-major, ld = DH
        const int kb3 = (lane >> 4) << 4;
#pragma unroll
        for (int dblk = 0; dblk < 4; ++dblk) {
            const __bf16* vp = vh + (size_t)(j + kb3) * DH_ + dblk * 16 + n;
            v16bf vb;
#pragma unroll
            for (int e = 0; e < 16; ++e) vb[e] = vp[(size_t)e * DH_];
            acc[dblk] = __builtin_amdgcn_wmma_f32_16x16x32_bf16(false, wa, false, vb,
                                                                (short)0, acc[dblk],
                                                                false, false);
        }
        __syncthreads();
    }
#pragma unroll
    for (int dblk = 0; dblk < 4; ++dblk) {
#pragma unroll
        for (int r = 0; r < 8; ++r) {
            const int mm  = mb + r;
            const int row = qt * 16 + mm;
            const int d   = dblk * 16 + n;
            const float rs  = __shfl(rowsum, mm, 32);
            const float val = acc[dblk][r] / (rs + 1e-6f) - vfh[(size_t)row * DH_ + d];
            mout[((size_t)(b * N_ + row)) * D_ + h * DH_ + d] = val;
        }
    }
}

// dst[t] = rmsnorm(a[t] + b[t]) over D=512. block = 256.
__global__ void k_addnorm(float* __restrict__ dst, const float* __restrict__ a,
                          const float* __restrict__ bsrc) {
    const int t   = blockIdx.x;
    const int tid = threadIdx.x;
    __shared__ float red[8];
    const size_t base = (size_t)t * D_;
    const float v0 = a[base + tid] + bsrc[base + tid];
    const float v1 = a[base + tid + 256] + bsrc[base + tid + 256];
    const float ss = wave_sum(v0 * v0 + v1 * v1);
    if ((tid & 31) == 0) red[tid >> 5] = ss;
    __syncthreads();
    float tot = 0.0f;
#pragma unroll
    for (int i = 0; i < 8; ++i) tot += red[i];
    const float inv = rsqrtf(tot * (1.0f / D_) + 1e-5f);
    dst[base + tid]       = v0 * inv;
    dst[base + tid + 256] = v1 * inv;
}

// Hf = silu(G) * U from GU[T][2*INNER]
__global__ void k_silugate(const float* __restrict__ GU, float* __restrict__ Hf) {
    const size_t i = (size_t)blockIdx.x * blockDim.x + threadIdx.x;
    if (i >= (size_t)T_ * INNER_) return;
    const size_t t = i / INNER_;
    const size_t c = i % INNER_;
    const float g = GU[t * (2 * INNER_) + c];
    const float u = GU[t * (2 * INNER_) + INNER_ + c];
    Hf[i] = silu(g) * u;
}

// Depthwise conv over sequence (KS=3, pad 1) + bias + silu
__global__ void k_dwconv(const float* __restrict__ Hf, const float* __restrict__ cw,
                         const float* __restrict__ cb, float* __restrict__ Hc) {
    const int c  = blockIdx.x * blockDim.x + threadIdx.x;
    const int nn = blockIdx.y;
    const int b  = blockIdx.z;
    if (c >= INNER_) return;
    float acc = cb[c];
#pragma unroll
    for (int s = 0; s < KS_; ++s) {
        const int mm = nn + s - 1;
        if (mm >= 0 && mm < N_)
            acc += cw[c * KS_ + s] * Hf[((size_t)b * N_ + mm) * INNER_ + c];
    }
    Hc[((size_t)b * N_ + nn) * INNER_ + c] = silu(acc);
}

// ---------------------------------------------------------------- launcher

extern "C" void kernel_launch(void* const* d_in, const int* in_sizes, int n_in,
                              void* d_out, int out_size, void* d_ws, size_t ws_size,
                              hipStream_t stream) {
    (void)in_sizes; (void)n_in; (void)out_size; (void)ws_size;
    const int*   ids   = (const int*)d_in[0];
    const float* emb   = (const float*)d_in[1];
    const float* ih    = (const float*)d_in[2];
    const float* Wqkv  = (const float*)d_in[3];
    const float* Wo    = (const float*)d_in[4];
    const float* Wup   = (const float*)d_in[5];
    const float* convw = (const float*)d_in[6];
    const float* convb = (const float*)d_in[7];
    const float* Wdn   = (const float*)d_in[8];
    const float* lmh   = (const float*)d_in[9];
    float* out = (float*)d_out;

    // ---------------- workspace: f32 region then bf16 region ----------------
    float* wsf = (float*)d_ws;
    size_t o = 0;
    float* X    = wsf + o; o += (size_t)T_ * D_;
    float* hid  = wsf + o; o += (size_t)T_ * D_;
    float* cosb = wsf + o; o += (size_t)N_ * DH_;
    float* sinb = wsf + o; o += (size_t)N_ * DH_;
    float* qkv  = wsf + o; o += (size_t)T_ * 3 * D_;   // reused as Hf
    float* vfb  = wsf + o; o += (size_t)T_ * D_;
    float* att  = wsf + o; o += (size_t)T_ * D_;
    float* tmp  = wsf + o; o += (size_t)T_ * D_;
    float* Qi   = wsf + o; o += (size_t)T_ * D_;
    float* GU   = wsf + o; o += (size_t)T_ * 2 * INNER_; // reused as Hc
    float* Hf   = qkv;   // overlay: qkv dead once q/k/v extracted
    float* Hc   = GU;    // overlay: GU dead once Hf computed

    __bf16* wsb = (__bf16*)(wsf + o);
    size_t ob = 0;
    __bf16* wbf_qkv = wsb + ob; ob += (size_t)L_ * 3 * D_ * D_;
    __bf16* wbf_o   = wsb + ob; ob += (size_t)L_ * D_ * D_;
    __bf16* wbf_up  = wsb + ob; ob += (size_t)L_ * 2 * INNER_ * D_;
    __bf16* wbf_dn  = wsb + ob; ob += (size_t)L_ * D_ * INNER_;
    __bf16* wbf_lmh = wsb + ob; ob += (size_t)V_ * D_;
    __bf16* abf     = wsb + ob; ob += (size_t)T_ * 3 * D_;  // activation staging
    __bf16* qbf     = wsb + ob; ob += (size_t)T_ * D_;
    __bf16* kbf     = wsb + ob; ob += (size_t)T_ * D_;
    __bf16* vbf     = wsb + ob; ob += (size_t)T_ * D_;

    // ---------------- one-time per launch: weights -> bf16 ----------------
    k_f2bf<<<(L_ * 3 * D_ * D_ + 255) / 256, 256, 0, stream>>>(Wqkv, wbf_qkv, L_ * 3 * D_ * D_);
    k_f2bf<<<(L_ * D_ * D_ + 255) / 256, 256, 0, stream>>>(Wo, wbf_o, L_ * D_ * D_);
    k_f2bf<<<(L_ * 2 * INNER_ * D_ + 255) / 256, 256, 0, stream>>>(Wup, wbf_up, L_ * 2 * INNER_ * D_);
    k_f2bf<<<(L_ * D_ * INNER_ + 255) / 256, 256, 0, stream>>>(Wdn, wbf_dn, L_ * D_ * INNER_);
    k_f2bf<<<(V_ * D_ + 255) / 256, 256, 0, stream>>>(lmh, wbf_lmh, V_ * D_);

    k_embed<<<(T_ * D_ + 255) / 256, 256, 0, stream>>>(ids, emb, ih, X, hid);
    k_ropecache<<<(N_ * DH_ + 255) / 256, 256, 0, stream>>>(cosb, sinb);

    // C = Af32[M,K] @ Wbf[Ncols,K]^T  (stage A to bf16, then WMMA GEMM)
    auto gemm = [&](const float* Af32, int M, int K, const __bf16* Wb, int Ncols,
                    float* Cout) {
        k_f2bf<<<(M * K + 255) / 256, 256, 0, stream>>>(Af32, abf, M * K);
        k_gemm_bf<<<dim3(Ncols / 256, M / 16), 128, 0, stream>>>(abf, Wb, Cout, Ncols, K);
    };

    for (int cyc = 0; cyc < CYC_; ++cyc) {
        k_add<<<(T_ * D_ + 255) / 256, 256, 0, stream>>>(hid, X, T_ * D_);
        for (int l = 0; l < L_; ++l) {
            gemm(hid, T_, D_, wbf_qkv + (size_t)l * 3 * D_ * D_, 3 * D_, qkv);
            k_ropeqkv<<<dim3(N_, H_, B_), 32, 0, stream>>>(qkv, cosb, sinb,
                                                           qbf, kbf, vbf, vfb);
            k_attn<<<dim3(N_ / 16, H_, B_), 32, 0, stream>>>(qbf, kbf, vbf, vfb, att);
            gemm(att, T_, D_, wbf_o + (size_t)l * D_ * D_, D_, tmp);
            k_addnorm<<<T_, 256, 0, stream>>>(Qi, hid, tmp);
            gemm(Qi, T_, D_, wbf_up + (size_t)l * 2 * INNER_ * D_, 2 * INNER_, GU);
            k_silugate<<<(T_ * INNER_ + 255) / 256, 256, 0, stream>>>(GU, Hf);
            k_dwconv<<<dim3(INNER_ / 256, N_, B_), 256, 0, stream>>>(
                Hf, convw + (size_t)l * INNER_ * KS_, convb + (size_t)l * INNER_, Hc);
            gemm(Hc, T_, INNER_, wbf_dn + (size_t)l * D_ * INNER_, D_, tmp);
            k_addnorm<<<T_, 256, 0, stream>>>(hid, Qi, tmp);
        }
    }
    gemm(hid, T_, D_, wbf_lmh, V_, out);
}
